// Attention_25520695673285
// MI455X (gfx1250) — compile-verified, matrix operand order checked by reading, not run on hardware
//
#include <hip/hip_runtime.h>

typedef __attribute__((ext_vector_type(16))) _Float16 v16h;
typedef __attribute__((ext_vector_type(8)))  _Float16 v8h;
typedef __attribute__((ext_vector_type(4)))  _Float16 v4h;
typedef __attribute__((ext_vector_type(8)))  float    v8f;

#define B_      16
#define S_      2048
#define H_      128
#define XT      128    // x rows per block (8 waves x 16)
#define YC      64     // y streamed per chunk
#define NW      8      // waves per block
#define THREADS 256

// Load a 16-half WMMA fragment from LDS: two contiguous 16B chunks at p and
// p+16 halves, matching the gfx1250 A/B per-lane K-interleave
// (lane<16: K {0..7,16..23}; lane>=16: K {8..15,24..31} via caller's hs*8).
__device__ __forceinline__ v16h ldfrag(const _Float16* p) {
    v8h a = *(const v8h*)(p);
    v8h b = *(const v8h*)(p + 16);
    return __builtin_shufflevector(a, b, 0, 1, 2, 3, 4, 5, 6, 7,
                                         8, 9, 10, 11, 12, 13, 14, 15);
}

// Global -> registers (f32 loads, immediate cvt to f16; only 32 VGPRs held).
__device__ __forceinline__ void load_chunk(const float* qb, const float* vb,
                                           int yc, int tid, int vh, int vy0,
                                           v4h qreg[8], v4h vreg[8]) {
    const float4* q4 = (const float4*)(qb + (size_t)yc * H_);
    #pragma unroll
    for (int i = 0; i < 8; ++i) {
        float4 qv = q4[tid + i * THREADS];
        qreg[i] = v4h{ (_Float16)qv.x, (_Float16)qv.y,
                       (_Float16)qv.z, (_Float16)qv.w };
    }
    const float* vc = vb + (size_t)yc * H_ + vh;   // column reads, coalesced/row
    #pragma unroll
    for (int i = 0; i < 8; ++i) {
        int y0 = vy0 + i * 4;
        vreg[i] = v4h{ (_Float16)vc[(y0 + 0) * H_], (_Float16)vc[(y0 + 1) * H_],
                       (_Float16)vc[(y0 + 2) * H_], (_Float16)vc[(y0 + 3) * H_] };
    }
}

// Registers -> LDS (Q row-major, V transposed), all 8-byte ds stores.
__device__ __forceinline__ void store_chunk(_Float16* bQ, _Float16* bVT,
                                            int tid, int vh, int vy0,
                                            const v4h qreg[8], const v4h vreg[8]) {
    #pragma unroll
    for (int i = 0; i < 8; ++i)
        *(v4h*)&bQ[4 * (tid + i * THREADS)] = qreg[i];
    #pragma unroll
    for (int i = 0; i < 8; ++i)
        *(v4h*)&bVT[vh * YC + vy0 + i * 4] = vreg[i];
}

__global__ __launch_bounds__(THREADS)
void fattn_rev_kernel(const float* __restrict__ qg, const float* __restrict__ kg,
                      const float* __restrict__ vg, float* __restrict__ og)
{
    __shared__ __align__(16) _Float16 sQd [2][YC * H_];   // [y][h] row-major
    __shared__ __align__(16) _Float16 sVTd[2][H_ * YC];   // [h][y] transposed
    __shared__ __align__(16) _Float16 sP  [NW * 16 * YC]; // per-wave P tile 16x64

    const int tid  = threadIdx.x;
    const int wave = tid >> 5;
    const int lane = tid & 31;
    const int ln16 = lane & 15;
    const int hs   = lane >> 4;                  // half-select for K interleave
    const int b    = blockIdx.x / (S_ / XT);
    const int x0   = (blockIdx.x % (S_ / XT)) * XT + wave * 16;

    // ---- K fragments (A-matrix 16x32 f16), one per 32-wide h-chunk ----
    v16h kf[4];
    {
        const float* kb = kg + ((size_t)b * S_ + (size_t)(x0 + ln16)) * H_;
        #pragma unroll
        for (int hc = 0; hc < 4; ++hc) {
            const float* p = kb + hc * 32 + hs * 8;
            #pragma unroll
            for (int j = 0; j < 8; ++j) {
                kf[hc][j]     = (_Float16)p[j];
                kf[hc][8 + j] = (_Float16)p[16 + j];
            }
        }
    }

    // all-ones B fragment: row sums of P via the matrix pipe (l = P x 1s)
    v16h ones;
    #pragma unroll
    for (int j = 0; j < 16; ++j) ones[j] = (_Float16)1.0f;

    // ---- running flash-attention state (per lane: 8 score rows) ----
    v8f zero = {};
    v8f acc[8];
    float mrun[8], lrun[8];
    #pragma unroll
    for (int i = 0; i < 8; ++i) { acc[i] = zero; mrun[i] = -1.0e30f; lrun[i] = 0.0f; }

    const float* qb = qg + (size_t)b * S_ * H_;
    const float* vb = vg + (size_t)b * S_ * H_;
    _Float16* sPw = &sP[wave * 16 * YC];

    const int vh  = tid & (H_ - 1);    // staging: this thread's V column (h)
    const int vy0 = (tid >> 7) * 32;   // staging: first y row (0 or 32)

    // ---- prologue: stage chunk 0 ----
    v4h qreg[8], vreg[8];
    load_chunk(qb, vb, 0, tid, vh, vy0, qreg, vreg);
    store_chunk(sQd[0], sVTd[0], tid, vh, vy0, qreg, vreg);
    __syncthreads();

    _Float16* curQ = sQd[0];
    _Float16* curV = sVTd[0];
    _Float16* nxtQ = sQd[1];
    _Float16* nxtV = sVTd[1];

    for (int yc = 0; yc < S_; yc += YC) {
        const bool more = (yc + YC < S_);
        if (more)                                  // regs for chunk i+1 in flight
            load_chunk(qb, vb, yc + YC, tid, vh, vy0, qreg, vreg);
        if (yc + 2 * YC < S_) {                    // chunk i+2 -> L2
            __builtin_prefetch(qb + (size_t)(yc + 2 * YC) * H_ + tid * 32, 0, 1);
            __builtin_prefetch(vb + (size_t)(yc + 2 * YC) * H_ + tid * 32, 0, 1);
        }

        // ---- scores: S[16x, 64y] as 4 column-fragments, K over h=128 ----
        v8f s0 = zero, s1 = zero, s2 = zero, s3 = zero;
        #pragma unroll
        for (int hc = 0; hc < 4; ++hc) {
            const int ho = hc * 32 + hs * 8;
            v16h bf0 = ldfrag(&curQ[( 0 + ln16) * H_ + ho]);
            v16h bf1 = ldfrag(&curQ[(16 + ln16) * H_ + ho]);
            v16h bf2 = ldfrag(&curQ[(32 + ln16) * H_ + ho]);
            v16h bf3 = ldfrag(&curQ[(48 + ln16) * H_ + ho]);
            s0 = __builtin_amdgcn_wmma_f32_16x16x32_f16(false, kf[hc], false, bf0,
                                                        (short)0, s0, false, false);
            s1 = __builtin_amdgcn_wmma_f32_16x16x32_f16(false, kf[hc], false, bf1,
                                                        (short)0, s1, false, false);
            s2 = __builtin_amdgcn_wmma_f32_16x16x32_f16(false, kf[hc], false, bf2,
                                                        (short)0, s2, false, false);
            s3 = __builtin_amdgcn_wmma_f32_16x16x32_f16(false, kf[hc], false, bf3,
                                                        (short)0, s3, false, false);
        }

        // ---- one online-softmax pass over all 64 y columns ----
        float alpha[8];
        #pragma unroll
        for (int r = 0; r < 8; ++r) {
            float t = fmaxf(fmaxf(s0[r], s1[r]), fmaxf(s2[r], s3[r]));
            t = fmaxf(t, __shfl_xor(t, 1));
            t = fmaxf(t, __shfl_xor(t, 2));
            t = fmaxf(t, __shfl_xor(t, 4));
            t = fmaxf(t, __shfl_xor(t, 8));       // row max within 16-lane half
            float mnew = fmaxf(mrun[r], t);
            alpha[r] = __expf(mrun[r] - mnew);
            mrun[r]  = mnew;
            float p0 = __expf(s0[r] - mnew);
            float p1 = __expf(s1[r] - mnew);
            float p2 = __expf(s2[r] - mnew);
            float p3 = __expf(s3[r] - mnew);
            int prow = hs ? (8 + r) : r;          // D-layout row for this lane
            _Float16* pr = &sPw[prow * YC + ln16];
            pr[0]  = (_Float16)p0;
            pr[16] = (_Float16)p1;
            pr[32] = (_Float16)p2;
            pr[48] = (_Float16)p3;
        }
        asm volatile("s_wait_dscnt 0" ::: "memory");  // wave-local LDS RAW

        // ---- P fragments (A-matrix 16x32 x2) ----
        v16h pf0 = ldfrag(&sPw[ln16 * YC +      hs * 8]);
        v16h pf1 = ldfrag(&sPw[ln16 * YC + 32 + hs * 8]);

        // ---- row sums on the matrix pipe: dz = P x ones ----
        v8f dz = zero;
        dz = __builtin_amdgcn_wmma_f32_16x16x32_f16(false, pf0, false, ones,
                                                    (short)0, dz, false, false);
        dz = __builtin_amdgcn_wmma_f32_16x16x32_f16(false, pf1, false, ones,
                                                    (short)0, dz, false, false);
        #pragma unroll
        for (int r = 0; r < 8; ++r) lrun[r] = lrun[r] * alpha[r] + dz[r];

        // ---- O = alpha*O + P(16x64) x V(64x16h) per 16-wide h-chunk ----
        #pragma unroll
        for (int h2 = 0; h2 < 8; ++h2) {
            const _Float16* vt = &curV[(h2 * 16 + ln16) * YC + hs * 8];
            v16h vf0 = ldfrag(vt);
            v16h vf1 = ldfrag(vt + 32);
            v8f o = acc[h2];
            #pragma unroll
            for (int r = 0; r < 8; ++r) o[r] *= alpha[r];
            o = __builtin_amdgcn_wmma_f32_16x16x32_f16(false, pf0, false, vf0,
                                                       (short)0, o, false, false);
            o = __builtin_amdgcn_wmma_f32_16x16x32_f16(false, pf1, false, vf1,
                                                       (short)0, o, false, false);
            acc[h2] = o;
        }

        // ---- publish chunk i+1 into the other buffer; single barrier ----
        if (more) {
            store_chunk(nxtQ, nxtV, tid, vh, vy0, qreg, vreg);
            __syncthreads();
            _Float16* t0 = curQ; curQ = nxtQ; nxtQ = t0;
            _Float16* t1 = curV; curV = nxtV; nxtV = t1;
        }
    }

    // ---- epilogue: divide by row sums, store f32 ----
    #pragma unroll
    for (int r = 0; r < 8; ++r) lrun[r] = 1.0f / lrun[r];
    float* ob = og + ((size_t)b * S_ + (size_t)x0) * H_;
    #pragma unroll
    for (int h2 = 0; h2 < 8; ++h2) {
        #pragma unroll
        for (int r = 0; r < 8; ++r) {
            int row = hs ? (8 + r) : r;
            ob[row * H_ + h2 * 16 + ln16] = acc[h2][r] * lrun[r];
        }
    }
}

extern "C" void kernel_launch(void* const* d_in, const int* in_sizes, int n_in,
                              void* d_out, int out_size, void* d_ws, size_t ws_size,
                              hipStream_t stream) {
    (void)in_sizes; (void)n_in; (void)d_ws; (void)ws_size; (void)out_size;
    const float* q = (const float*)d_in[0];
    const float* k = (const float*)d_in[1];
    const float* v = (const float*)d_in[2];
    float* out = (float*)d_out;
    dim3 grid(B_ * (S_ / XT));   // 256 blocks
    dim3 block(THREADS);         // 8 waves (wave32)
    hipLaunchKernelGGL(fattn_rev_kernel, grid, block, 0, stream, q, k, v, out);
}
